// SetTransformer_58669253263514
// MI455X (gfx1250) — matrix operand order repaired and checked by reading, not compiled, not run on hardware
//
#include <hip/hip_runtime.h>
#include <math.h>

// ---------------- types for CDNA5 WMMA ----------------
typedef __bf16 bf16_t;
typedef __attribute__((ext_vector_type(16))) __bf16 v16bf;
typedef __attribute__((ext_vector_type(8)))  __bf16 bf16x8;
typedef __attribute__((ext_vector_type(8)))  float  v8f;

#define N_NODES 16384
#define D_FIL   512
#define D_IN    532
#define D_PAD   544      // 532 padded to 17*32 for K-steps of 32
#define D_MODEL 1024
#define NHEAD   16
#define D_KEY   64
#define DFF     4096
#define T_TREES 128
#define EPSLN   1e-5f

__device__ inline v8f wmma_bf16(v16bf a, v16bf b, v8f c) {
  return __builtin_amdgcn_wmma_f32_16x16x32_bf16(false, a, false, b, (short)0, c, false, false);
}

// A-fragment (16x32 bf16, row-major source, ISA 16-bit A layout):
// lane<16 : row=lane, elems0-7=K0..7,  elems8-15=K16..23
// lane>=16: row=lane-16, elems0-7=K8..15, elems8-15=K24..31
__device__ inline v16bf load_a_frag(const bf16_t* A, int lda, int lane) {
  int row = lane & 15, half = lane >> 4;
  const bf16_t* p = A + row * lda + half * 8;
  union { bf16x8 h[2]; v16bf v; } u;
  u.h[0] = *reinterpret_cast<const bf16x8*>(p);
  u.h[1] = *reinterpret_cast<const bf16x8*>(p + 16);
  return u.v;
}

// B-fragment (32x16 bf16).  B[k][c] = W[c][k], W row-major with ldb.
// lane<16 : col=lane, elems0-7=K0..7, elems8-15=K16..23 ; lane>=16: col=lane-16, K8..15, K24..31
__device__ inline v16bf load_b_frag(const bf16_t* W, int ldb, int lane) {
  int col = lane & 15, half = lane >> 4;
  const bf16_t* p = W + col * ldb + half * 8;
  union { bf16x8 h[2]; v16bf v; } u;
  u.h[0] = *reinterpret_cast<const bf16x8*>(p);
  u.h[1] = *reinterpret_cast<const bf16x8*>(p + 16);
  return u.v;
}

// ---------------- K0a: generic f32 -> bf16 (with K padding) ----------------
__global__ void conv_pad_bf16(const float* src, bf16_t* dst, int rows, int scols, int dcols) {
  int i = blockIdx.x * 256 + threadIdx.x;
  if (i >= rows * dcols) return;
  int r = i / dcols, c = i - r * dcols;
  dst[i] = (c < scols) ? (bf16_t)src[r * scols + c] : (bf16_t)0.0f;
}

// ---------------- K0b: Wq_eff[h,k] = sum_d q[h,d]*Wk[h*64+d, k]  (bf16, padded) -----
__global__ void wqeff_kernel(const float* q, const float* wk, bf16_t* dst) {
  int i = blockIdx.x * 256 + threadIdx.x;
  if (i >= NHEAD * D_PAD) return;
  int h = i / D_PAD, k = i - h * D_PAD;
  float s = 0.0f;
  if (k < D_IN) {
    const float* qr = q + h * D_KEY;
    const float* wr = wk + (h * D_KEY) * D_IN + k;
    for (int d = 0; d < D_KEY; ++d) s += qr[d] * wr[d * D_IN];
  }
  dst[i] = (bf16_t)s;
}

// ---------------- K1: per-node LN(values) ++ tree-pos-enc, LN again -> x_bf --------
__global__ void node_stage(const float* __restrict__ values, const int* __restrict__ role,
                           const float* g_fil, const float* b_fil,
                           const float* g_mha, const float* b_mha,
                           bf16_t* __restrict__ x_bf) {
  int n = blockIdx.x, tid = threadIdx.x;
  __shared__ float s1[256], s2[256];
  const float* vr = values + (size_t)n * D_FIL;
  float v0 = vr[tid], v1 = vr[tid + 256];
  s1[tid] = v0 + v1; s2[tid] = v0 * v0 + v1 * v1;
  __syncthreads();
  for (int off = 128; off > 0; off >>= 1) {
    if (tid < off) { s1[tid] += s1[tid + off]; s2[tid] += s2[tid + off]; }
    __syncthreads();
  }
  float mu = s1[0] * (1.0f / D_FIL);
  float var = s2[0] * (1.0f / D_FIL) - mu * mu;
  float rs = rsqrtf(var + EPSLN);
  __syncthreads();
  float f0 = (v0 - mu) * rs * g_fil[tid] + b_fil[tid];
  float f1 = (v1 - mu) * rs * g_fil[tid + 256] + b_fil[tid + 256];
  float tp = 0.0f;
  if (tid < 20) {
    int r = role[n];
    int cm = 31 - __builtin_clz((unsigned)r);        // highest set bit
    int bit = (r >> tid) & 1;
    tp = (tid < cm) ? (bit ? 1.0f : -1.0f) : 0.0f;   // {0,1}->{-1,1} below MSB, MSB->0
  }
  s1[tid] = f0 + f1 + tp; s2[tid] = f0 * f0 + f1 * f1 + tp * tp;
  __syncthreads();
  for (int off = 128; off > 0; off >>= 1) {
    if (tid < off) { s1[tid] += s1[tid + off]; s2[tid] += s2[tid + off]; }
    __syncthreads();
  }
  float mu2 = s1[0] * (1.0f / D_IN);
  float var2 = s2[0] * (1.0f / D_IN) - mu2 * mu2;
  float rs2 = rsqrtf(var2 + EPSLN);
  bf16_t* xr = x_bf + (size_t)n * D_PAD;
  xr[tid]       = (bf16_t)((f0 - mu2) * rs2 * g_mha[tid] + b_mha[tid]);
  xr[tid + 256] = (bf16_t)((f1 - mu2) * rs2 * g_mha[tid + 256] + b_mha[tid + 256]);
  if (tid < 20) xr[512 + tid] = (bf16_t)((tp - mu2) * rs2 * g_mha[512 + tid] + b_mha[512 + tid]);
  if (tid < 12) xr[532 + tid] = (bf16_t)0.0f;  // zero pad for GEMM K
}

// ---------------- K2: qkm = x @ Wq_eff.T  (WMMA), pre-scaled by 1/sqrt(64) ---------
__global__ void qkm_gemm(const bf16_t* __restrict__ x_bf, const bf16_t* __restrict__ wqe,
                         float* __restrict__ qkm) {
  int lane = threadIdx.x & 31;
  int m0 = blockIdx.x * 16;
  v8f acc = {};
  const bf16_t* Abase = x_bf + (size_t)m0 * D_PAD;
  for (int kt = 0; kt < D_PAD / 32; ++kt) {
    v16bf a = load_a_frag(Abase + kt * 32, D_PAD, lane);
    v16bf b = load_b_frag(wqe + kt * 32, D_PAD, lane);
    acc = wmma_bf16(a, b, acc);
  }
  int col = lane & 15, half = lane >> 4;
  for (int i = 0; i < 8; ++i)
    qkm[(size_t)(m0 + half * 8 + i) * NHEAD + col] = acc[i] * 0.125f;
}

// ---------------- K3: deterministic per-tree node lists ---------------------------
__global__ void zero_cnt(int* c) { if (threadIdx.x < T_TREES) c[threadIdx.x] = 0; }

__global__ void scatter_nodes(const int* bi, const int* mi, int* cnt, int* idx) {
  int n = blockIdx.x * 256 + threadIdx.x;
  if (n >= N_NODES) return;
  int t = bi[n] * 8 + mi[n];
  int p = atomicAdd(&cnt[t], 1);
  idx[t * T_TREES + p] = n;
}

__global__ void sort128(int* idx) {   // bitonic sort -> deterministic fp order
  __shared__ int s[128];
  int t = threadIdx.x;
  int* row = idx + blockIdx.x * T_TREES;
  s[t] = row[t]; __syncthreads();
  for (int k = 2; k <= 128; k <<= 1)
    for (int j = k >> 1; j > 0; j >>= 1) {
      int ixj = t ^ j;
      if (ixj > t) {
        bool up = (t & k) == 0;
        int a = s[t], b = s[ixj];
        if ((a > b) == up) { s[t] = b; s[ixj] = a; }
      }
      __syncthreads();
    }
  row[t] = s[t];
}

// ---------------- K4: softmax over tree + xbar[t,h,:] = sum_n w[n,h]*x[n,:] -------
__global__ void softmax_xbar(const int* __restrict__ idx, const float* __restrict__ qkm,
                             const bf16_t* __restrict__ x_bf, bf16_t* __restrict__ xbar) {
  int t = blockIdx.x, tid = threadIdx.x;
  __shared__ int   il[128];
  __shared__ float wl[NHEAD * 128];
  if (tid < 128) il[tid] = idx[t * T_TREES + tid];
  __syncthreads();
  for (int i = tid; i < NHEAD * 128; i += 256) {
    int h = i >> 7, n = i & 127;
    wl[h * 128 + n] = qkm[(size_t)il[n] * NHEAD + h];  // already scaled by 1/8
  }
  __syncthreads();
  if (tid < NHEAD) {
    float* w = wl + tid * 128;
    float mx = -3.4e38f;
    for (int n = 0; n < 128; ++n) mx = fmaxf(mx, w[n]);
    float s = 0.0f;
    for (int n = 0; n < 128; ++n) { float e = expf(w[n] - mx); w[n] = e; s += e; }
    float inv = 1.0f / s;
    for (int n = 0; n < 128; ++n) w[n] *= inv;
  }
  __syncthreads();
  int h = tid >> 4, kb = tid & 15;
  float acc[34];
  for (int j = 0; j < 34; ++j) acc[j] = 0.0f;
  const float* w = wl + h * 128;
  for (int n = 0; n < 128; ++n) {
    const bf16_t* xr = x_bf + (size_t)il[n] * D_PAD;
    float wn = w[n];
    for (int j = 0; j < 34; ++j) acc[j] += wn * (float)xr[kb + 16 * j];
  }
  bf16_t* o = xbar + ((size_t)t * NHEAD + h) * D_PAD;
  for (int j = 0; j < 34; ++j) o[kb + 16 * j] = (bf16_t)acc[j];
}

// ---------------- K5: attn out[t, h*64+d] = xbar[t,h,:] . Wv[h*64+d,:]  (WMMA) -----
__global__ void attn_out_gemm(const bf16_t* __restrict__ xbar, const bf16_t* __restrict__ wv,
                              bf16_t* __restrict__ attn_bf) {
  int lane = threadIdx.x & 31;
  int dt = blockIdx.x, tt = blockIdx.y, h = blockIdx.z;
  const bf16_t* Abase = xbar + (size_t)tt * 16 * (NHEAD * D_PAD) + (size_t)h * D_PAD;
  const bf16_t* Bbase = wv + (size_t)(h * D_KEY + dt * 16) * D_PAD;
  v8f acc = {};
  for (int kt = 0; kt < D_PAD / 32; ++kt) {
    v16bf a = load_a_frag(Abase + kt * 32, NHEAD * D_PAD, lane);
    v16bf b = load_b_frag(Bbase + kt * 32, D_PAD, lane);
    acc = wmma_bf16(a, b, acc);
  }
  int col = lane & 15, half = lane >> 4;
  for (int i = 0; i < 8; ++i) {
    int tr = tt * 16 + half * 8 + i;
    int d  = h * D_KEY + dt * 16 + col;
    attn_bf[(size_t)tr * D_MODEL + d] = (bf16_t)acc[i];
  }
}

// ---------------- K6: r1 = attn @ Wo.T + bo + q_flat  (WMMA) ----------------------
__global__ void r1_gemm(const bf16_t* __restrict__ attn_bf, const bf16_t* __restrict__ wo,
                        const float* __restrict__ bo, const float* __restrict__ qflat,
                        float* __restrict__ r1) {
  int lane = threadIdx.x & 31;
  int nt = blockIdx.x, tt = blockIdx.y;
  const bf16_t* Abase = attn_bf + (size_t)tt * 16 * D_MODEL;
  const bf16_t* Bbase = wo + (size_t)nt * 16 * D_MODEL;
  v8f acc = {};
  for (int kt = 0; kt < D_MODEL / 32; ++kt) {
    v16bf a = load_a_frag(Abase + kt * 32, D_MODEL, lane);
    v16bf b = load_b_frag(Bbase + kt * 32, D_MODEL, lane);
    acc = wmma_bf16(a, b, acc);
  }
  int col = lane & 15, half = lane >> 4;
  int nc = nt * 16 + col;
  float add = bo[nc] + qflat[nc];
  for (int i = 0; i < 8; ++i) {
    int tr = tt * 16 + half * 8 + i;
    r1[(size_t)tr * D_MODEL + nc] = acc[i] + add;
  }
}

// ---------------- LN over 1024 cols -> bf16 ---------------------------------------
__global__ void ln1024_to_bf16(const float* __restrict__ in, const float* g, const float* b,
                               bf16_t* __restrict__ out) {
  int r = blockIdx.x, tid = threadIdx.x;
  __shared__ float s1[256], s2[256];
  const float* row = in + (size_t)r * D_MODEL;
  float x[4], ps = 0.0f, pq = 0.0f;
  for (int j = 0; j < 4; ++j) { x[j] = row[tid + 256 * j]; ps += x[j]; pq += x[j] * x[j]; }
  s1[tid] = ps; s2[tid] = pq; __syncthreads();
  for (int off = 128; off > 0; off >>= 1) {
    if (tid < off) { s1[tid] += s1[tid + off]; s2[tid] += s2[tid + off]; }
    __syncthreads();
  }
  float mu = s1[0] * (1.0f / D_MODEL);
  float var = s2[0] * (1.0f / D_MODEL) - mu * mu;
  float rs = rsqrtf(var + EPSLN);
  for (int j = 0; j < 4; ++j) {
    int c = tid + 256 * j;
    out[(size_t)r * D_MODEL + c] = (bf16_t)((x[j] - mu) * rs * g[c] + b[c]);
  }
}

// ---------------- K8: g = gelu(l1 @ W1.T + b1)  (WMMA) ----------------------------
__global__ void ffn1_gemm(const bf16_t* __restrict__ l1, const bf16_t* __restrict__ w1,
                          const float* __restrict__ b1, bf16_t* __restrict__ gout) {
  int lane = threadIdx.x & 31;
  int nt = blockIdx.x, tt = blockIdx.y;
  const bf16_t* Abase = l1 + (size_t)tt * 16 * D_MODEL;
  const bf16_t* Bbase = w1 + (size_t)nt * 16 * D_MODEL;
  v8f acc = {};
  for (int kt = 0; kt < D_MODEL / 32; ++kt) {
    __builtin_prefetch(Bbase + (kt + 1) * 32, 0, 0);
    v16bf a = load_a_frag(Abase + kt * 32, D_MODEL, lane);
    v16bf b = load_b_frag(Bbase + kt * 32, D_MODEL, lane);
    acc = wmma_bf16(a, b, acc);
  }
  int col = lane & 15, half = lane >> 4;
  int nc = nt * 16 + col;
  float bb = b1[nc];
  for (int i = 0; i < 8; ++i) {
    int tr = tt * 16 + half * 8 + i;
    float z = acc[i] + bb;
    float gl = 0.5f * z * (1.0f + erff(z * 0.70710678118654752f));  // exact gelu
    gout[(size_t)tr * DFF + nc] = (bf16_t)gl;
  }
}

// ---------------- K9: ff = g @ W2.T + b2 + r1  (WMMA) -----------------------------
__global__ void ffn2_gemm(const bf16_t* __restrict__ gin, const bf16_t* __restrict__ w2,
                          const float* __restrict__ b2, const float* __restrict__ r1,
                          float* __restrict__ ff) {
  int lane = threadIdx.x & 31;
  int nt = blockIdx.x, tt = blockIdx.y;
  const bf16_t* Abase = gin + (size_t)tt * 16 * DFF;
  const bf16_t* Bbase = w2 + (size_t)nt * 16 * DFF;
  v8f acc = {};
  for (int kt = 0; kt < DFF / 32; ++kt) {
    __builtin_prefetch(Bbase + (kt + 1) * 32, 0, 0);
    v16bf a = load_a_frag(Abase + kt * 32, DFF, lane);
    v16bf b = load_b_frag(Bbase + kt * 32, DFF, lane);
    acc = wmma_bf16(a, b, acc);
  }
  int col = lane & 15, half = lane >> 4;
  int nc = nt * 16 + col;
  float bb = b2[nc];
  for (int i = 0; i < 8; ++i) {
    int tr = tt * 16 + half * 8 + i;
    ff[(size_t)tr * D_MODEL + nc] = acc[i] + bb + r1[(size_t)tr * D_MODEL + nc];
  }
}

// ---------------- K10: final LN -> d_out (+ zero agent_pad_mask) ------------------
__global__ void final_ln(const float* __restrict__ in, const float* g, const float* b,
                         float* __restrict__ out) {
  int r = blockIdx.x, tid = threadIdx.x;
  __shared__ float s1[256], s2[256];
  const float* row = in + (size_t)r * D_MODEL;
  float x[4], ps = 0.0f, pq = 0.0f;
  for (int j = 0; j < 4; ++j) { x[j] = row[tid + 256 * j]; ps += x[j]; pq += x[j] * x[j]; }
  s1[tid] = ps; s2[tid] = pq; __syncthreads();
  for (int off = 128; off > 0; off >>= 1) {
    if (tid < off) { s1[tid] += s1[tid + off]; s2[tid] += s2[tid + off]; }
    __syncthreads();
  }
  float mu = s1[0] * (1.0f / D_MODEL);
  float var = s2[0] * (1.0f / D_MODEL) - mu * mu;
  float rs = rsqrtf(var + EPSLN);
  for (int j = 0; j < 4; ++j) {
    int c = tid + 256 * j;
    out[(size_t)r * D_MODEL + c] = (x[j] - mu) * rs * g[c] + b[c];
  }
  if (r == 0 && tid < T_TREES) out[(size_t)T_TREES * D_MODEL + tid] = 0.0f;  // pad mask
}

// ---------------- launch ----------------------------------------------------------
extern "C" void kernel_launch(void* const* d_in, const int* in_sizes, int n_in,
                              void* d_out, int out_size, void* d_ws, size_t ws_size,
                              hipStream_t stream) {
  const float* values = (const float*)d_in[0];
  const int*   role   = (const int*)d_in[1];
  const int*   bidx   = (const int*)d_in[2];
  const int*   midx   = (const int*)d_in[3];
  const float* qvec   = (const float*)d_in[4];
  const float* Wk     = (const float*)d_in[5];
  const float* Wv     = (const float*)d_in[6];
  const float* Wo     = (const float*)d_in[7];
  const float* bo     = (const float*)d_in[8];
  const float* W1     = (const float*)d_in[9];
  const float* b1     = (const float*)d_in[10];
  const float* W2     = (const float*)d_in[11];
  const float* b2     = (const float*)d_in[12];
  const float* g_fil  = (const float*)d_in[13];
  const float* b_fil  = (const float*)d_in[14];
  const float* g_mha  = (const float*)d_in[15];
  const float* b_mha  = (const float*)d_in[16];
  const float* g_ff   = (const float*)d_in[17];
  const float* b_ff   = (const float*)d_in[18];
  const float* g_out  = (const float*)d_in[19];
  const float* b_out  = (const float*)d_in[20];

  char* ws = (char*)d_ws;
  size_t off = 0;
  auto carve = [&](size_t bytes) { char* p = ws + off; off += (bytes + 255) & ~(size_t)255; return p; };
  bf16_t* x_bf    = (bf16_t*)carve((size_t)N_NODES * D_PAD * 2);
  bf16_t* wqe_bf  = (bf16_t*)carve((size_t)NHEAD * D_PAD * 2);
  bf16_t* wv_bf   = (bf16_t*)carve((size_t)D_MODEL * D_PAD * 2);
  bf16_t* wo_bf   = (bf16_t*)carve((size_t)D_MODEL * D_MODEL * 2);
  bf16_t* w1_bf   = (bf16_t*)carve((size_t)DFF * D_MODEL * 2);
  bf16_t* w2_bf   = (bf16_t*)carve((size_t)D_MODEL * DFF * 2);
  float*  qkm     = (float*)carve((size_t)N_NODES * NHEAD * 4);
  int*    tidx    = (int*)carve((size_t)T_TREES * T_TREES * 4);
  int*    tcnt    = (int*)carve((size_t)T_TREES * 4);
  bf16_t* xbar    = (bf16_t*)carve((size_t)T_TREES * NHEAD * D_PAD * 2);
  bf16_t* attn_bf = (bf16_t*)carve((size_t)T_TREES * D_MODEL * 2);
  float*  r1      = (float*)carve((size_t)T_TREES * D_MODEL * 4);
  bf16_t* l1_bf   = (bf16_t*)carve((size_t)T_TREES * D_MODEL * 2);
  bf16_t* g_bf    = (bf16_t*)carve((size_t)T_TREES * DFF * 2);
  float*  ff      = (float*)carve((size_t)T_TREES * D_MODEL * 4);

  // weight prep (bf16, K-padded)
  conv_pad_bf16<<<(D_MODEL * D_PAD + 255) / 256, 256, 0, stream>>>(Wv, wv_bf, D_MODEL, D_IN, D_PAD);
  conv_pad_bf16<<<(D_MODEL * D_MODEL + 255) / 256, 256, 0, stream>>>(Wo, wo_bf, D_MODEL, D_MODEL, D_MODEL);
  conv_pad_bf16<<<(DFF * D_MODEL + 255) / 256, 256, 0, stream>>>(W1, w1_bf, DFF, D_MODEL, D_MODEL);
  conv_pad_bf16<<<(D_MODEL * DFF + 255) / 256, 256, 0, stream>>>(W2, w2_bf, D_MODEL, DFF, DFF);
  wqeff_kernel<<<(NHEAD * D_PAD + 255) / 256, 256, 0, stream>>>(qvec, Wk, wqe_bf);

  // node stage: LN + tree positional encodings + LN -> bf16 x
  node_stage<<<N_NODES, 256, 0, stream>>>(values, role, g_fil, b_fil, g_mha, b_mha, x_bf);

  // qkm = x @ Wq_eff.T  (folded q into Wk; scaled by 1/sqrt(D_KEY))
  qkm_gemm<<<N_NODES / 16, 32, 0, stream>>>(x_bf, wqe_bf, qkm);

  // deterministic per-tree node lists
  zero_cnt<<<1, 128, 0, stream>>>(tcnt);
  scatter_nodes<<<N_NODES / 256, 256, 0, stream>>>(bidx, midx, tcnt, tidx);
  sort128<<<T_TREES, 128, 0, stream>>>(tidx);

  // softmax over each tree + attention-weighted x reduction
  softmax_xbar<<<T_TREES, 256, 0, stream>>>(tidx, qkm, x_bf, xbar);

  // out[t,:] = xbar @ Wv.T (per head), then Wo + residual, FFN, LNs
  attn_out_gemm<<<dim3(D_KEY / 16, T_TREES / 16, NHEAD), 32, 0, stream>>>(xbar, wv_bf, attn_bf);
  r1_gemm<<<dim3(D_MODEL / 16, T_TREES / 16), 32, 0, stream>>>(attn_bf, wo_bf, bo, qvec, r1);
  ln1024_to_bf16<<<T_TREES, 256, 0, stream>>>(r1, g_ff, b_ff, l1_bf);
  ffn1_gemm<<<dim3(DFF / 16, T_TREES / 16), 32, 0, stream>>>(l1_bf, w1_bf, b1, g_bf);
  ffn2_gemm<<<dim3(D_MODEL / 16, T_TREES / 16), 32, 0, stream>>>(g_bf, w2_bf, b2, r1, ff);
  final_ln<<<T_TREES, 256, 0, stream>>>(ff, g_out, b_out, (float*)d_out);
}
